// TransformerBlock1DWithAttn_60662118089032
// MI455X (gfx1250) — compile-verified, matrix operand order checked by reading, not run on hardware
//
#include <hip/hip_runtime.h>

// ---------------------------------------------------------------------------
// CDNA5 (gfx1250) wave32 WMMA transformer block.
// ---------------------------------------------------------------------------
typedef __attribute__((ext_vector_type(16))) __bf16 bf16x16;
typedef __attribute__((ext_vector_type(8)))  float  f32x8;
typedef __attribute__((ext_vector_type(4)))  int    i32x4;

union Frag16 { bf16x16 v; uint4 q[2]; };
union Pack8  { uint4 q; __bf16 h[8]; };

#define AS1 __attribute__((address_space(1)))
#define AS3 __attribute__((address_space(3)))

#if defined(__gfx1250__) && __has_builtin(__builtin_amdgcn_global_load_async_to_lds_b128) && __has_builtin(__builtin_amdgcn_s_wait_asynccnt)
#define USE_ASYNC_LDS 1
#else
#define USE_ASYNC_LDS 0
#endif

// 16-byte global -> LDS copy (async DMA path on CDNA5, VGPR bounce otherwise)
__device__ inline void cp_b128(const __bf16* g, __bf16* l) {
#if USE_ASYNC_LDS
    __builtin_amdgcn_global_load_async_to_lds_b128(
        (AS1 i32x4*)(AS1 void*)g, (AS3 i32x4*)(AS3 void*)l, 0, 0);
#else
    *(uint4*)l = *(const uint4*)g;
#endif
}
__device__ inline void cp_join() {
#if USE_ASYNC_LDS
    __builtin_amdgcn_s_wait_asynccnt(0);
#endif
}

__device__ inline f32x8 zero8() {
    f32x8 z;
#pragma unroll
    for (int i = 0; i < 8; ++i) z[i] = 0.0f;
    return z;
}

// ---------------------------------------------------------------------------
// Tiled fp32->bf16 transpose: out[n*K + k] = (bf16) in[k*N + n]
// One-time per weight matrix; makes every GEMM B-operand K-contiguous.
// ---------------------------------------------------------------------------
__global__ __launch_bounds__(256) void transpose_f32_bf16(
    const float* __restrict__ in, __bf16* __restrict__ out, int K, int N) {
    __shared__ float tile[32][33];
    const int tx = threadIdx.x;        // 0..31
    const int ty = threadIdx.y;        // 0..7
    const int k0 = blockIdx.y * 32;
    const int n0 = blockIdx.x * 32;
#pragma unroll
    for (int i = 0; i < 4; ++i)
        tile[ty + 8 * i][tx] = in[(size_t)(k0 + ty + 8 * i) * N + n0 + tx];
    __syncthreads();
#pragma unroll
    for (int i = 0; i < 4; ++i)
        out[(size_t)(n0 + ty + 8 * i) * K + k0 + tx] = (__bf16)tile[tx][ty + 8 * i];
}

// ---------------------------------------------------------------------------
// Row LayerNorm (C = 1024), fp32 in -> bf16 out. One block per row.
// ---------------------------------------------------------------------------
__global__ __launch_bounds__(256) void layernorm_bf16(const float* __restrict__ x,
                                                      const float* __restrict__ gamma,
                                                      const float* __restrict__ beta,
                                                      __bf16* __restrict__ y, int C) {
    __shared__ float red[256];
    const int row = blockIdx.x;
    const int tid = threadIdx.x;
    const float* xr = x + (size_t)row * C;

    float s = 0.0f, s2 = 0.0f;
    for (int i = tid; i < C; i += 256) { float v = xr[i]; s += v; s2 += v * v; }

    red[tid] = s; __syncthreads();
    for (int off = 128; off > 0; off >>= 1) { if (tid < off) red[tid] += red[tid + off]; __syncthreads(); }
    const float mean = red[0] / (float)C;
    __syncthreads();
    red[tid] = s2; __syncthreads();
    for (int off = 128; off > 0; off >>= 1) { if (tid < off) red[tid] += red[tid + off]; __syncthreads(); }
    const float var = red[0] / (float)C - mean * mean;
    const float rstd = rsqrtf(var + 1e-5f);

    __bf16* yr = y + (size_t)row * C;
    for (int i = tid; i < C; i += 256)
        yr[i] = (__bf16)((xr[i] - mean) * rstd * gamma[i] + beta[i]);
}

// ---------------------------------------------------------------------------
// Tiled WMMA GEMM: C[M,N] = A[M,K] * B[K,N] (+bias) (+relu) (+residual)
// A row-major bf16 [M,K]; BT = B^T row-major bf16 [N,K] (pre-transposed).
// Block tile 128x128, K-step 32, 256 threads = 8 waves (2 M x 4 N),
// each wave 64x32 = 4x2 WMMA tiles -> 8 v_wmma per K-step.
// ---------------------------------------------------------------------------
template <bool RELU>
__global__ __launch_bounds__(256) void gemm_bf16_wmma(
    const __bf16* __restrict__ A, const __bf16* __restrict__ BT,
    const float* __restrict__ bias, const float* __restrict__ resid,
    float* __restrict__ outF, __bf16* __restrict__ outB,
    int M, int N, int K) {
    constexpr int BM = 128, BN = 128, BK = 32, STR = BK + 8;
    __shared__ __bf16 As[BM][STR];   // A tile  [m][k]
    __shared__ __bf16 Bs[BN][STR];   // B^T tile [n][k]

    const int tid = threadIdx.x;
    const int lane = tid & 31;
    const int wave = tid >> 5;
    const int wm = wave >> 2;            // 0..1
    const int wn = wave & 3;             // 0..3
    const int l16 = lane & 15;
    const int halfsel = lane >> 4;       // 0/1
    const int kfrag = halfsel * 8;       // K chunks: {kfrag..+7, kfrag+16..+23}
    const int m0 = blockIdx.y * BM;
    const int n0 = blockIdx.x * BN;

    f32x8 acc[4][2];
#pragma unroll
    for (int i = 0; i < 4; ++i)
#pragma unroll
        for (int j = 0; j < 2; ++j) acc[i][j] = zero8();

    for (int k0 = 0; k0 < K; k0 += BK) {
        // --- global -> LDS: both tiles are 128x32, K-contiguous ---
#pragma unroll
        for (int it = 0; it < 2; ++it) {
            int idx = (tid + it * 256) * 8;       // 0..4088
            int r = idx >> 5, c = idx & 31;
            cp_b128(A  + (size_t)(m0 + r) * K + k0 + c, &As[r][c]);
            cp_b128(BT + (size_t)(n0 + r) * K + k0 + c, &Bs[r][c]);
        }
        if (k0 + BK < K) {  // hint next K-tile into cache (global_prefetch_b8)
            __builtin_prefetch(A  + (size_t)(m0 + (tid >> 1)) * K + k0 + BK, 0, 1);
            __builtin_prefetch(BT + (size_t)(n0 + (tid >> 1)) * K + k0 + BK, 0, 1);
        }
        cp_join();
        __syncthreads();

        // --- fragments + 8 WMMAs per wave ---
        Frag16 bf[2];
#pragma unroll
        for (int tn = 0; tn < 2; ++tn) {
            int n = wn * 32 + tn * 16 + l16;
            bf[tn].q[0] = *(const uint4*)(&Bs[n][kfrag]);
            bf[tn].q[1] = *(const uint4*)(&Bs[n][kfrag + 16]);
        }
#pragma unroll
        for (int tm = 0; tm < 4; ++tm) {
            int r = wm * 64 + tm * 16 + l16;
            Frag16 af;
            af.q[0] = *(const uint4*)(&As[r][kfrag]);
            af.q[1] = *(const uint4*)(&As[r][kfrag + 16]);
#pragma unroll
            for (int tn = 0; tn < 2; ++tn) {
                acc[tm][tn] = __builtin_amdgcn_wmma_f32_16x16x32_bf16(
                    false, af.v, false, bf[tn].v, (short)0, acc[tm][tn], false, false);
            }
        }
        __syncthreads();
    }

    // --- epilogue: bias / relu / residual, fp32 and/or bf16 out ---
#pragma unroll
    for (int tm = 0; tm < 4; ++tm) {
#pragma unroll
        for (int tn = 0; tn < 2; ++tn) {
            const int col = n0 + wn * 32 + tn * 16 + l16;
            const float bv = bias ? bias[col] : 0.0f;
#pragma unroll
            for (int r = 0; r < 8; ++r) {
                const int row = m0 + wm * 64 + tm * 16 + halfsel * 8 + r;
                float v = acc[tm][tn][r] + bv;
                if (RELU) v = v > 0.0f ? v : 0.0f;
                const size_t o = (size_t)row * N + col;
                if (resid) v += resid[o];
                if (outF) outF[o] = v;
                if (outB) outB[o] = (__bf16)v;
            }
        }
    }
}

// ---------------------------------------------------------------------------
// Flash attention, one (b, h, 64-query tile) per block. 256 thr = 8 waves.
// qkv layout: [B*L][3*C] with q@0, k@1024, v@2048, head h at h*64.
// Output: attn[B*L][1024], head h at columns h*64..h*64+63 (pre out-proj).
// ---------------------------------------------------------------------------
__global__ __launch_bounds__(256) void flash_attn_bf16(
    const __bf16* __restrict__ qkv, __bf16* __restrict__ attn,
    int L, int H, float scale) {
    constexpr int Dh = 64, BQ = 64, BKV = 64, STR = 72, SSTR = 68, QKVW = 3072, C = 1024;
    __shared__ __bf16 qs[BQ][STR];
    __shared__ __bf16 ks[BKV][STR];
    __shared__ __bf16 vt[Dh][STR];    // vt[d][key]
    __shared__ __bf16 ps[BQ][STR];    // probabilities (bf16)
    __shared__ float  ss[BQ][SSTR];   // raw logits (fp32)
    __shared__ float  mrow[BQ], lrow[BQ], arow[BQ];

    const int tid = threadIdx.x;
    const int lane = tid & 31;
    const int wave = tid >> 5;
    const int wq = wave >> 1;          // 0..3 : 16-row query strip
    const int wx = wave & 1;           // 0..1 : key-half (S) / dh-half (O)
    const int l16 = lane & 15;
    const int halfsel = lane >> 4;
    const int kfrag = halfsel * 8;

    const int bh = blockIdx.y;
    const int b = bh / H, h = bh % H;
    const int q0 = blockIdx.x * BQ;
    const size_t tokbase = (size_t)b * L;
    const int qcol = h * Dh, kcol = C + h * Dh, vcol = 2 * C + h * Dh;

    const int tr = tid >> 2;           // 0..63 tile row for copies
    const int tc = (tid & 3) * 16;     // 0/16/32/48 tile col

    // load Q tile (64x64) via async DMA
    cp_b128(qkv + (tokbase + q0 + tr) * QKVW + qcol + tc,     &qs[tr][tc]);
    cp_b128(qkv + (tokbase + q0 + tr) * QKVW + qcol + tc + 8, &qs[tr][tc + 8]);
    cp_join();
    if (tid < BQ) { mrow[tid] = -1e30f; lrow[tid] = 0.0f; }

    f32x8 oacc[2];
    oacc[0] = zero8(); oacc[1] = zero8();

    for (int j0 = 0; j0 < L; j0 += BKV) {
        __syncthreads();  // prev iteration's P@V done; also covers Q/stat init
        // K tile via async DMA; V tile transposed through VGPRs
        cp_b128(qkv + (tokbase + j0 + tr) * QKVW + kcol + tc,     &ks[tr][tc]);
        cp_b128(qkv + (tokbase + j0 + tr) * QKVW + kcol + tc + 8, &ks[tr][tc + 8]);
        {
            const uint4* srcV = (const uint4*)(qkv + (tokbase + j0 + tr) * QKVW + vcol + tc);
            Pack8 p0, p1;
            p0.q = srcV[0]; p1.q = srcV[1];
#pragma unroll
            for (int e = 0; e < 8; ++e) { vt[tc + e][tr] = p0.h[e]; vt[tc + 8 + e][tr] = p1.h[e]; }
        }
        cp_join();
        __syncthreads();

        // --- S = Q K^T : wave computes rows wq*16..+15, cols wx*32..+31 ---
        f32x8 sacc[2];
        sacc[0] = zero8(); sacc[1] = zero8();
#pragma unroll
        for (int kc = 0; kc < Dh; kc += 32) {
            const int rq = wq * 16 + l16;
            Frag16 aq;
            aq.q[0] = *(const uint4*)(&qs[rq][kc + kfrag]);
            aq.q[1] = *(const uint4*)(&qs[rq][kc + kfrag + 16]);
#pragma unroll
            for (int tn = 0; tn < 2; ++tn) {
                const int nk = wx * 32 + tn * 16 + l16;
                Frag16 bk;
                bk.q[0] = *(const uint4*)(&ks[nk][kc + kfrag]);
                bk.q[1] = *(const uint4*)(&ks[nk][kc + kfrag + 16]);
                sacc[tn] = __builtin_amdgcn_wmma_f32_16x16x32_bf16(
                    false, aq.v, false, bk.v, (short)0, sacc[tn], false, false);
            }
        }
        // spill S to LDS
#pragma unroll
        for (int tn = 0; tn < 2; ++tn)
#pragma unroll
            for (int r = 0; r < 8; ++r)
                ss[wq * 16 + halfsel * 8 + r][wx * 32 + tn * 16 + l16] = sacc[tn][r];
        __syncthreads();

        // --- online softmax per row (threads 0..63) ---
        if (tid < BQ) {
            const int r = tid;
            float mj = -1e30f;
            for (int c2 = 0; c2 < BKV; ++c2) {
                float v = ss[r][c2] * scale; ss[r][c2] = v;
                mj = fmaxf(mj, v);
            }
            const float mprev = mrow[r];
            const float mn = fmaxf(mprev, mj);
            const float al = __expf(mprev - mn);
            float sum = 0.0f;
            for (int c2 = 0; c2 < BKV; ++c2) {
                float p = __expf(ss[r][c2] - mn);
                ps[r][c2] = (__bf16)p;
                sum += p;
            }
            lrow[r] = lrow[r] * al + sum;
            mrow[r] = mn;
            arow[r] = al;
        }
        __syncthreads();

        // --- rescale O and accumulate P @ V ---
#pragma unroll
        for (int tn = 0; tn < 2; ++tn)
#pragma unroll
            for (int r = 0; r < 8; ++r)
                oacc[tn][r] *= arow[wq * 16 + halfsel * 8 + r];
#pragma unroll
        for (int kc = 0; kc < BKV; kc += 32) {
            const int rq = wq * 16 + l16;
            Frag16 ap;
            ap.q[0] = *(const uint4*)(&ps[rq][kc + kfrag]);
            ap.q[1] = *(const uint4*)(&ps[rq][kc + kfrag + 16]);
#pragma unroll
            for (int tn = 0; tn < 2; ++tn) {
                const int nd = wx * 32 + tn * 16 + l16;
                Frag16 bv;
                bv.q[0] = *(const uint4*)(&vt[nd][kc + kfrag]);
                bv.q[1] = *(const uint4*)(&vt[nd][kc + kfrag + 16]);
                oacc[tn] = __builtin_amdgcn_wmma_f32_16x16x32_bf16(
                    false, ap.v, false, bv.v, (short)0, oacc[tn], false, false);
            }
        }
    }

    // --- finalize: O / l, write bf16 into attn[b*L+q][h*64 + d] ---
#pragma unroll
    for (int tn = 0; tn < 2; ++tn) {
#pragma unroll
        for (int r = 0; r < 8; ++r) {
            const int rloc = wq * 16 + halfsel * 8 + r;
            const float linv = 1.0f / lrow[rloc];
            const int col = h * Dh + wx * 32 + tn * 16 + l16;
            attn[(tokbase + q0 + rloc) * C + col] = (__bf16)(oacc[tn][r] * linv);
        }
    }
}

// ---------------------------------------------------------------------------
// Host launch
// ---------------------------------------------------------------------------
extern "C" void kernel_launch(void* const* d_in, const int* in_sizes, int n_in,
                              void* d_out, int out_size, void* d_ws, size_t ws_size,
                              hipStream_t stream) {
    (void)in_sizes; (void)n_in; (void)out_size; (void)ws_size;
    const float* src    = (const float*)d_in[0];
    const float* w_qkv  = (const float*)d_in[1];
    const float* w_out  = (const float*)d_in[2];
    const float* b_out  = (const float*)d_in[3];
    const float* w1     = (const float*)d_in[4];
    const float* b1     = (const float*)d_in[5];
    const float* w2     = (const float*)d_in[6];
    const float* b2     = (const float*)d_in[7];
    const float* gamma1 = (const float*)d_in[8];
    const float* beta1  = (const float*)d_in[9];
    const float* gamma2 = (const float*)d_in[10];
    const float* beta2  = (const float*)d_in[11];
    float* out = (float*)d_out;

    constexpr int B = 4, L = 2048, C = 1024, H = 16, Dh = 64, FF = 4096;
    constexpr int M = B * L;  // 8192 tokens
    const float scale = 0.125f;  // Dh^-0.5

    char* ws = (char*)d_ws;
    size_t off = 0;
    auto alloc = [&](size_t bytes) -> void* {
        void* p = ws + off;
        off += (bytes + 255) & ~(size_t)255;
        return p;
    };
    // pre-transposed bf16 weights: [N][K]
    __bf16* wqkvT = (__bf16*)alloc((size_t)3 * C * C * 2);   // 6 MB
    __bf16* woutT = (__bf16*)alloc((size_t)C * C * 2);       // 2 MB
    __bf16* w1T   = (__bf16*)alloc((size_t)FF * C * 2);      // 8 MB
    __bf16* w2T   = (__bf16*)alloc((size_t)C * FF * 2);      // 8 MB
    __bf16* xn_bf   = (__bf16*)alloc((size_t)M * C * 2);     // 16 MB (xn1 then xn2)
    __bf16* qkv_bf  = (__bf16*)alloc((size_t)M * 3 * C * 2); // 48 MB
    __bf16* attn_bf = (__bf16*)alloc((size_t)M * C * 2);     // 16 MB
    float*  src2_f  = (float*)alloc((size_t)M * C * 4);      // 32 MB
    // FFN hidden aliases the (dead) qkv+attn regions: 64 MB needed, 64 MB there.
    __bf16* h_bf = qkv_bf;

    // 1) weights -> bf16, transposed to [N][K]
    dim3 tb(32, 8);
    transpose_f32_bf16<<<dim3(3 * C / 32, C / 32), tb, 0, stream>>>(w_qkv, wqkvT, C, 3 * C);
    transpose_f32_bf16<<<dim3(C / 32, C / 32), tb, 0, stream>>>(w_out, woutT, C, C);
    transpose_f32_bf16<<<dim3(FF / 32, C / 32), tb, 0, stream>>>(w1, w1T, C, FF);
    transpose_f32_bf16<<<dim3(C / 32, FF / 32), tb, 0, stream>>>(w2, w2T, FF, C);

    // 2) xn1 = LN(src)
    layernorm_bf16<<<M, 256, 0, stream>>>(src, gamma1, beta1, xn_bf, C);

    // 3) qkv = xn1 @ w_qkv   [8192 x 3072]
    gemm_bf16_wmma<false><<<dim3(3 * C / 128, M / 128), 256, 0, stream>>>(
        xn_bf, wqkvT, nullptr, nullptr, nullptr, qkv_bf, M, 3 * C, C);

    // 4) flash attention -> attn_bf [8192 x 1024]
    flash_attn_bf16<<<dim3(L / 64, B * H), 256, 0, stream>>>(qkv_bf, attn_bf, L, H, scale);

    // 5) src2 = src + attn @ w_out + b_out   (fp32)
    gemm_bf16_wmma<false><<<dim3(C / 128, M / 128), 256, 0, stream>>>(
        attn_bf, woutT, b_out, src, src2_f, nullptr, M, C, C);

    // 6) xn2 = LN(src2)
    layernorm_bf16<<<M, 256, 0, stream>>>(src2_f, gamma2, beta2, xn_bf, C);

    // 7) h = relu(xn2 @ w1 + b1)   [8192 x 4096], bf16 (aliases qkv region)
    gemm_bf16_wmma<true><<<dim3(FF / 128, M / 128), 256, 0, stream>>>(
        xn_bf, w1T, b1, nullptr, nullptr, h_bf, M, FF, C);

    // 8) out = src2 + h @ w2 + b2   (fp32, final)
    gemm_bf16_wmma<false><<<dim3(C / 128, M / 128), 256, 0, stream>>>(
        h_bf, w2T, b2, src2_f, out, nullptr, M, C, FF);
}
